// STRG_36017595744856
// MI455X (gfx1250) — matrix operand order repaired and checked by reading, not compiled, not run on hardware
//
#include <hip/hip_runtime.h>
#include <hip/hip_bf16.h>

typedef __attribute__((ext_vector_type(16))) _Float16 v16h;
typedef __attribute__((ext_vector_type(8)))  _Float16 v8h;
typedef __attribute__((ext_vector_type(8)))  float    v8f;

#define NN      50000
#define NFEAT   512
#define NHID    128
#define NCLASS  40
#define NCPAD   48
#define NEDGE   1600000
#define MTILES  3125        // NN/16

// ---------------- scratch layout (bytes) ----------------
#define OFF_XW1 ((size_t)0)
#define OFF_S1  (OFF_XW1 + (size_t)NN*NHID*4)
#define OFF_H   (OFF_S1  + (size_t)NN*NHID*4)
#define OFF_W1P (OFF_H   + (size_t)NN*NHID*2)
#define OFF_W2P (OFF_W1P + (size_t)NFEAT*NHID*2)
#define OFF_HW2 (OFF_W2P + (size_t)NHID*NCPAD*2)
#define OFF_S2  (OFF_HW2 + (size_t)NN*NCPAD*4)

// ---------------- utility: zero fill ----------------
__global__ void zero_f32(float* __restrict__ p, int n) {
    int i = blockIdx.x * blockDim.x + threadIdx.x;
    if (i < n) p[i] = 0.0f;
}

// ---------------- pack W1 into WMMA B-fragment order ----------------
// w1p[((nt*16 + kt)*32 + lane)*16 + e] = W1[k][n]
//   n = nt*16 + lane%16 ; k = kt*32 + (e/8)*16 + (lane/16)*8 + e%8
__global__ void pack_w1(const float* __restrict__ W1, _Float16* __restrict__ w1p) {
    int t = blockIdx.x * blockDim.x + threadIdx.x;     // 8*16*32*16 = 65536
    int e  = t & 15;
    int l  = (t >> 4) & 31;
    int kt = (t >> 9) & 15;
    int nt = t >> 13;
    int n = nt * 16 + (l & 15);
    int k = kt * 32 + (e >> 3) * 16 + (l >> 4) * 8 + (e & 7);
    w1p[t] = (_Float16)W1[(size_t)k * NHID + n];
}

// ---------------- pack W2 (pad 40 -> 48 cols) ----------------
__global__ void pack_w2(const float* __restrict__ W2, _Float16* __restrict__ w2p) {
    int t = blockIdx.x * blockDim.x + threadIdx.x;     // 3*4*32*16 = 6144
    int e  = t & 15;
    int l  = (t >> 4) & 31;
    int kt = (t >> 9) & 3;
    int nt = t >> 11;
    int n = nt * 16 + (l & 15);                        // 0..47
    int k = kt * 32 + (e >> 3) * 16 + (l >> 4) * 8 + (e & 7);  // 0..127
    float v = (n < NCLASS) ? W2[(size_t)k * NCLASS + n] : 0.0f;
    w2p[t] = (_Float16)v;
}

// ---------------- GEMM1: XW1 = x(f32, cvt f16) @ W1 ----------------
// one wave per 16-row slab; 8 accumulators cover all of NHID=128.
// A fragment built once per K-step and reused across all 8 N-tiles,
// so x is streamed from HBM exactly once.
__global__ __launch_bounds__(128) void gemm1_wmma(const float* __restrict__ x,
                                                  const _Float16* __restrict__ w1p,
                                                  float* __restrict__ xw1) {
    const int wave = threadIdx.x >> 5;
    const int lane = threadIdx.x & 31;
    const int mt   = blockIdx.x * 4 + wave;
    if (mt >= MTILES) return;                          // uniform per wave
    const int g = lane >> 4;
    const int m = (mt << 4) + (lane & 15);

    const float*    arow  = x + (size_t)m * NFEAT;
    const _Float16* bbase = w1p + (size_t)lane * 16;   // + ((nt*16+kt)*32)*16

    v8f c[8] = {};
    for (int kt = 0; kt < 16; ++kt) {
        const int kb = kt * 32;
        float4 a0 = *(const float4*)(arow + kb + 8 * g);
        float4 a1 = *(const float4*)(arow + kb + 8 * g + 4);
        float4 a2 = *(const float4*)(arow + kb + 16 + 8 * g);
        float4 a3 = *(const float4*)(arow + kb + 16 + 8 * g + 4);
        v16h a;
        a[0]=(_Float16)a0.x; a[1]=(_Float16)a0.y; a[2]=(_Float16)a0.z; a[3]=(_Float16)a0.w;
        a[4]=(_Float16)a1.x; a[5]=(_Float16)a1.y; a[6]=(_Float16)a1.z; a[7]=(_Float16)a1.w;
        a[8]=(_Float16)a2.x; a[9]=(_Float16)a2.y; a[10]=(_Float16)a2.z; a[11]=(_Float16)a2.w;
        a[12]=(_Float16)a3.x; a[13]=(_Float16)a3.y; a[14]=(_Float16)a3.z; a[15]=(_Float16)a3.w;
        #pragma unroll
        for (int nt = 0; nt < 8; ++nt) {
            v16h b = *(const v16h*)(bbase + ((size_t)(nt * 16 + kt)) * 512);
            c[nt] = __builtin_amdgcn_wmma_f32_16x16x32_f16(false, a, false, b,
                                                           (short)0, c[nt], false, false);
        }
    }
    float* obase = xw1 + (size_t)((mt << 4) + 8 * g) * NHID + (lane & 15);
    #pragma unroll
    for (int nt = 0; nt < 8; ++nt) {
        #pragma unroll
        for (int v = 0; v < 8; ++v)
            obase[(size_t)v * NHID + (nt << 4)] = c[nt][v];
    }
}

// ---------------- SpMM1: s1[dst] += val * xw1[src]  (128 feats) ----------------
// one wave per edge (wave-uniform edge data), 4 feats per lane
__global__ __launch_bounds__(256) void spmm1(const int* __restrict__ src,
                                             const int* __restrict__ dst,
                                             const float* __restrict__ val,
                                             const float* __restrict__ xw1,
                                             float* __restrict__ s1) {
    const int e    = blockIdx.x * 8 + (threadIdx.x >> 5);  // NEDGE exact
    const int lane = threadIdx.x & 31;
    const int s = src[e];
    const int d = dst[e];
    const float w = val[e];
    const float4 v = *(const float4*)(xw1 + (size_t)s * NHID + lane * 4);
    float* o = s1 + (size_t)d * NHID + lane * 4;
    atomicAdd(o + 0, w * v.x);
    atomicAdd(o + 1, w * v.y);
    atomicAdd(o + 2, w * v.z);
    atomicAdd(o + 3, w * v.w);
}

// ---------------- h = relu(s1 + b1) -> f16 ----------------
__global__ void relu_bias_h(const float* __restrict__ s1, const float* __restrict__ b1,
                            _Float16* __restrict__ h) {
    int i = blockIdx.x * blockDim.x + threadIdx.x;     // NN*NHID exact
    float v = s1[i] + b1[i & (NHID - 1)];
    h[i] = (_Float16)(v > 0.0f ? v : 0.0f);
}

// ---------------- GEMM2: HW2 = h(f16) @ W2pad -> NN x 48 f32 ----------------
// one wave per 16-row slab; 3 accumulators cover NCPAD=48.
__global__ __launch_bounds__(128) void gemm2_wmma(const _Float16* __restrict__ h,
                                                  const _Float16* __restrict__ w2p,
                                                  float* __restrict__ hw2) {
    const int wave = threadIdx.x >> 5;
    const int lane = threadIdx.x & 31;
    const int mt   = blockIdx.x * 4 + wave;
    if (mt >= MTILES) return;                          // uniform per wave
    const int g = lane >> 4;
    const int m = (mt << 4) + (lane & 15);

    const _Float16* arow  = h + (size_t)m * NHID;
    const _Float16* bbase = w2p + (size_t)lane * 16;   // + ((nt*4+kt)*32)*16

    v8f c[3] = {};
    #pragma unroll
    for (int kt = 0; kt < 4; ++kt) {
        const int kb = kt * 32;
        v8h lo = *(const v8h*)(arow + kb + 8 * g);
        v8h hi = *(const v8h*)(arow + kb + 16 + 8 * g);
        v16h a;
        #pragma unroll
        for (int i = 0; i < 8; ++i) { a[i] = lo[i]; a[8 + i] = hi[i]; }
        #pragma unroll
        for (int nt = 0; nt < 3; ++nt) {
            v16h b = *(const v16h*)(bbase + ((size_t)(nt * 4 + kt)) * 512);
            c[nt] = __builtin_amdgcn_wmma_f32_16x16x32_f16(false, a, false, b,
                                                           (short)0, c[nt], false, false);
        }
    }
    float* obase = hw2 + (size_t)((mt << 4) + 8 * g) * NCPAD + (lane & 15);
    #pragma unroll
    for (int nt = 0; nt < 3; ++nt) {
        #pragma unroll
        for (int v = 0; v < 8; ++v)
            obase[(size_t)v * NCPAD + (nt << 4)] = c[nt][v];
    }
}

// ---------------- SpMM2: s2[dst] += val * hw2[src]  (40 feats) ----------------
__global__ __launch_bounds__(256) void spmm2(const int* __restrict__ src,
                                             const int* __restrict__ dst,
                                             const float* __restrict__ val,
                                             const float* __restrict__ hw2,
                                             float* __restrict__ s2) {
    const int e    = blockIdx.x * 8 + (threadIdx.x >> 5);
    const int lane = threadIdx.x & 31;
    const int s = src[e];
    const int d = dst[e];
    const float w = val[e];
    const int f = lane * 2;
    if (f < NCLASS) {
        float2 v = *(const float2*)(hw2 + (size_t)s * NCPAD + f);
        atomicAdd(s2 + (size_t)d * NCLASS + f,     w * v.x);
        atomicAdd(s2 + (size_t)d * NCLASS + f + 1, w * v.y);
    }
}

// ---------------- log_softmax(s2 + b2) ----------------
__global__ void logsoftmax_rows(const float* __restrict__ s2, const float* __restrict__ b2,
                                float* __restrict__ out) {
    int r = blockIdx.x * blockDim.x + threadIdx.x;
    if (r >= NN) return;
    float vals[NCLASS];
    float mx = -1e30f;
    #pragma unroll
    for (int c = 0; c < NCLASS; ++c) {
        float v = s2[(size_t)r * NCLASS + c] + b2[c];
        vals[c] = v;
        mx = v > mx ? v : mx;
    }
    float sum = 0.0f;
    #pragma unroll
    for (int c = 0; c < NCLASS; ++c) sum += __expf(vals[c] - mx);
    float lz = mx + __logf(sum);
    #pragma unroll
    for (int c = 0; c < NCLASS; ++c) out[(size_t)r * NCLASS + c] = vals[c] - lz;
}

extern "C" void kernel_launch(void* const* d_in, const int* in_sizes, int n_in,
                              void* d_out, int out_size, void* d_ws, size_t ws_size,
                              hipStream_t stream) {
    const float* x    = (const float*)d_in[0];
    const float* W1   = (const float*)d_in[1];
    const float* b1   = (const float*)d_in[2];
    const float* W2   = (const float*)d_in[3];
    const float* b2   = (const float*)d_in[4];
    const int*   esrc = (const int*)d_in[5];
    const int*   edst = (const int*)d_in[6];
    const float* eval = (const float*)d_in[7];
    float* out = (float*)d_out;

    char* ws = (char*)d_ws;
    float*    xw1 = (float*)   (ws + OFF_XW1);
    float*    s1  = (float*)   (ws + OFF_S1);
    _Float16* h   = (_Float16*)(ws + OFF_H);
    _Float16* w1p = (_Float16*)(ws + OFF_W1P);
    _Float16* w2p = (_Float16*)(ws + OFF_W2P);
    float*    hw2 = (float*)   (ws + OFF_HW2);
    float*    s2  = (float*)   (ws + OFF_S2);

    // zero the atomic accumulators (every call — graph-safe, deterministic)
    zero_f32<<<(NN * NHID + 255) / 256, 256, 0, stream>>>(s1, NN * NHID);
    zero_f32<<<(NN * NCLASS + 255) / 256, 256, 0, stream>>>(s2, NN * NCLASS);

    // pack weights into WMMA fragment order
    pack_w1<<<65536 / 256, 256, 0, stream>>>(W1, w1p);
    pack_w2<<<6144 / 256, 256, 0, stream>>>(W2, w2p);

    // layer 1
    gemm1_wmma<<<(MTILES + 3) / 4, 128, 0, stream>>>(x, w1p, xw1);
    spmm1<<<NEDGE / 8, 256, 0, stream>>>(esrc, edst, eval, xw1, s1);
    relu_bias_h<<<(NN * NHID) / 256, 256, 0, stream>>>(s1, b1, h);

    // layer 2
    gemm2_wmma<<<(MTILES + 3) / 4, 128, 0, stream>>>(h, w2p, hw2);
    spmm2<<<NEDGE / 8, 256, 0, stream>>>(esrc, edst, eval, hw2, s2);

    // log-softmax
    logsoftmax_rows<<<(NN + 255) / 256, 256, 0, stream>>>(s2, b2, out);
}
